// FramePriorNetwork_48619029791406
// MI455X (gfx1250) — matrix operand order repaired and checked by reading, not compiled, not run on hardware
//
#include <hip/hip_runtime.h>
#include <hip/hip_bf16.h>

// ---------------- problem constants (from reference) ----------------
#define H     512
#define FF    2048
#define NH    8
#define NL    6
#define KS    3
#define WIN   4
#define HEAD  64          // H / NH
#define BATCH 4
#define T     800
#define NTOT  (BATCH * T)
#define BHT   ((size_t)BATCH * H * T)

typedef _Float16 v16h __attribute__((ext_vector_type(16)));
typedef float    v8f  __attribute__((ext_vector_type(8)));
typedef int      v4i  __attribute__((vector_size(4 * sizeof(int))));

// -------- CDNA5 async global->LDS DMA (probe-guarded, device pass only) ----
#if defined(__AMDGCN__) && defined(__has_builtin)
#if __has_builtin(__builtin_amdgcn_global_load_async_to_lds_b128)
#define USE_ASYNC_LDS 1
#endif
#endif

// param0 of the async builtin is as(1) v4i* (clang prints it as "__device__"),
// param1 is the LDS-side pointer -> as(3) v4i*.
#define GV4I(p) ((__attribute__((address_space(1))) v4i*)(p))
#define LV4I(p) ((__attribute__((address_space(3))) v4i*)(p))

__device__ __forceinline__ void asyncWait() {
#ifdef USE_ASYNC_LDS
#if __has_builtin(__builtin_amdgcn_s_wait_asynccnt)
  __builtin_amdgcn_s_wait_asynccnt(0);
#else
  asm volatile("s_wait_asynccnt 0" ::: "memory");
#endif
#endif
}

__device__ __forceinline__ v8f wmma_f16(v16h a, v16h b, v8f c) {
  // D = A(16x32 f16) x B(32x16 f16) + C(16x16 f32)
  return __builtin_amdgcn_wmma_f32_16x16x32_f16(false, a, false, b, (short)0, c,
                                                false, false);
}

// K-quarter swizzle so each lane's 16 B-halves are one contiguous 32B ds read.
__device__ __forceinline__ int kswz(int kk) {  // kk in [0,32)
  int q = kk >> 3;
  return (q == 1 || q == 2) ? (kk ^ 24) : kk;
}

// ======================================================================
// Weight prep: src [M,K,TAPS] f32  ->  dst [TAPS][M][K] f16
// ======================================================================
__global__ void k_w2h(const float* __restrict__ src, _Float16* __restrict__ dst,
                      int M, int K, int TAPS) {
  size_t i = (size_t)blockIdx.x * 256 + threadIdx.x;  // over TAPS*M*K
  size_t total = (size_t)TAPS * M * K;
  if (i >= total) return;
  int k = (int)(i % K);
  size_t mm = i / K;
  int m = (int)(mm % M);
  int tap = (int)(mm / M);
  dst[i] = (_Float16)src[((size_t)m * K + k) * TAPS + tap];
}

// ======================================================================
// WMMA GEMM:  Y[M,N] = act( W16[tap][M][K] (x) X[K,N] + bias )
//   N = BATCH*T columns in [B,K,T] layout. TAPS=3 = the KS=3 conv.
// Block: 256 thr = 8 waves, tile 64(M) x 128(N); wave owns 32x32 = 2x2
// fragments -> 4 WMMAs/step. Double-buffered LDS:
//   A tile (f16 weights): one global_load_async_to_lds_b128 per thread,
//     overlapped with this step's WMMAs, settled by s_wait_asynccnt+barrier.
//   B tile: register-pipelined f32 load + cvt + transpose (mask/tap/edge
//     logic lives here); the cvt VALU co-executes with the XDL WMMAs.
// ======================================================================
template <int TAPS, bool RELU>
__global__ __launch_bounds__(256) void k_gemm(const _Float16* __restrict__ W16,
                                              const float* __restrict__ bias,
                                              const float* __restrict__ X,
                                              const float* __restrict__ colmask,
                                              float* __restrict__ Y, int M, int K) {
  alignas(32) __shared__ _Float16 As[2][64][32];   // [m][k] raw async tile
  alignas(32) __shared__ _Float16 Bs[2][128][48];  // [n][k] staged transposed

  const int tid  = threadIdx.x;
  const int lane = tid & 31;
  const int wave = tid >> 5;
  const int wy = wave >> 2;   // 0..1 -> 32-row M sub-tile
  const int wx = wave & 3;    // 0..3 -> 32-col N sub-tile
  const int mBase = blockIdx.x * 64;
  const int nBase = blockIdx.y * 128;

  const int kSteps = K >> 5;
  const int nsteps = kSteps * TAPS;

  float bN[16];  // next B tile: 32(K)x128(N), 16 per thread (coalesced in t)

  auto copyA = [&](int s, int dbuf) {  // 64x32 f16 = 4KB = 256 x b128
    const int tap = s / kSteps;
    const int k0  = (s % kSteps) << 5;
#ifdef USE_ASYNC_LDS
    const int m = tid >> 2, k8 = (tid & 3) << 3;
    const _Float16* src = W16 + ((size_t)tap * M + mBase + m) * K + k0 + k8;
    __builtin_amdgcn_global_load_async_to_lds_b128(
        GV4I(src), LV4I(&As[dbuf][m][k8]), 0, 0);
#else
#pragma unroll
    for (int j = 0; j < 8; ++j) {
      int idx = j * 256 + tid;
      int m = idx >> 5, kk = idx & 31;
      As[dbuf][m][kk] = W16[((size_t)tap * M + mBase + m) * K + k0 + kk];
    }
#endif
  };
  auto loadB = [&](int s) {
    const int tap = s / kSteps;
    const int k0  = (s % kSteps) << 5;
    const int shift = tap - (TAPS >> 1);
#pragma unroll
    for (int j = 0; j < 16; ++j) {
      int idx = j * 256 + tid;
      int n = idx & 127, kk = idx >> 7;  // lanes walk N(t): contiguous
      int col = nBase + n;
      int b = col / T, t = col % T;
      int ts = t + shift;
      float xv = 0.f;
      if (ts >= 0 && ts < T) {
        xv = X[((size_t)b * K + kk + k0) * T + ts];
        if (colmask) xv *= colmask[b * T + ts];
      }
      bN[j] = xv;
    }
  };
  auto storeB = [&](int dbuf) {
#pragma unroll
    for (int j = 0; j < 16; ++j) {
      int idx = j * 256 + tid;
      int n = idx & 127, kk = idx >> 7;
      Bs[dbuf][n][kk] = (_Float16)bN[j];
    }
  };

  v8f acc[2][2] = {};

  copyA(0, 0);
  loadB(0);
  storeB(0);
  asyncWait();
  __syncthreads();

  int db = 0;
  for (int s = 0; s < nsteps; ++s) {
    if (s + 1 < nsteps) {
      copyA(s + 1, db ^ 1);  // async DMA overlaps the WMMAs below
      loadB(s + 1);
    }
    if (s + 1 < nsteps)      // speculative L2 warm for next X rows
      __builtin_prefetch(&X[(size_t)(((s + 1) % kSteps) << 5) * T + nBase % T], 0, 1);
    {
      const int row  = lane & 15;
      const int koff = (lane >> 4) << 4;
      v16h a0 = *(const v16h*)&As[db][wy * 32 + row][koff];
      v16h a1 = *(const v16h*)&As[db][wy * 32 + 16 + row][koff];
      v16h b0 = *(const v16h*)&Bs[db][wx * 32 + row][koff];
      v16h b1 = *(const v16h*)&Bs[db][wx * 32 + 16 + row][koff];
      acc[0][0] = wmma_f16(a0, b0, acc[0][0]);
      acc[0][1] = wmma_f16(a0, b1, acc[0][1]);
      acc[1][0] = wmma_f16(a1, b0, acc[1][0]);
      acc[1][1] = wmma_f16(a1, b1, acc[1][1]);
    }
    if (s + 1 < nsteps) storeB(db ^ 1);
    asyncWait();
    __syncthreads();
    db ^= 1;
  }

  const int hi = lane >> 4, nloc = lane & 15;
#pragma unroll
  for (int mf = 0; mf < 2; ++mf)
#pragma unroll
    for (int nf = 0; nf < 2; ++nf)
#pragma unroll
      for (int r = 0; r < 8; ++r) {
        int m = mBase + wy * 32 + mf * 16 + r + hi * 8;
        int n = nBase + wx * 32 + nf * 16 + nloc;
        int b = n / T, t = n % T;
        float val = acc[mf][nf][r] + bias[m];
        if (RELU) val = val > 0.f ? val : 0.f;
        Y[((size_t)b * M + m) * T + t] = val;
      }
}

// ======================================================================
// scores[b,h,t,s] = scale * sum_d q[d,t] k[d,s]   (one wave per 16x16 tile)
// ======================================================================
__global__ __launch_bounds__(32) void k_scores(const float* __restrict__ q,
                                               const float* __restrict__ kmat,
                                               float* __restrict__ scores) {
  alignas(32) __shared__ _Float16 Aq[16][80];  // [t][d]
  alignas(32) __shared__ _Float16 Bk[16][80];  // [s][d]
  const int lane = threadIdx.x;
  const int bh = blockIdx.z;
  const int b = bh / NH, hh = bh % NH;
  const int tBase = blockIdx.x * 16, sBase = blockIdx.y * 16;
  const float scale = 0.125f;  // HEAD^-0.5
  const size_t coff = ((size_t)b * H + hh * HEAD) * T;

#pragma unroll 4
  for (int j = 0; j < 32; ++j) {
    int idx = j * 32 + lane;  // 16 rows x 64 d
    int r = idx >> 6, d = idx & 63;
    int dsw = (d & 32) | kswz(d & 31);
    Aq[r][dsw] = (_Float16)(q[coff + (size_t)d * T + tBase + r] * scale);
    Bk[r][dsw] = (_Float16)(kmat[coff + (size_t)d * T + sBase + r]);
  }
  __syncthreads();

  v8f acc = {};
  const int row = lane & 15, koff = (lane >> 4) << 4;
#pragma unroll
  for (int kk = 0; kk < HEAD; kk += 32) {
    v16h a  = *(const v16h*)&Aq[row][kk + koff];
    v16h bf = *(const v16h*)&Bk[row][kk + koff];
    acc = wmma_f16(a, bf, acc);
  }
  const int hi = lane >> 4, nloc = lane & 15;
#pragma unroll
  for (int r = 0; r < 8; ++r) {
    int tt = tBase + r + hi * 8;
    scores[((size_t)bh * T + tt) * T + sBase + nloc] = acc[r];
  }
}

// ======================================================================
// Fused: scores += band-limited rel-k logits (|s-t|<=WIN), mask, softmax.
// ======================================================================
__global__ __launch_bounds__(256) void k_softmax(float* __restrict__ scores,
                                                 const float* __restrict__ q,
                                                 const float* __restrict__ ek,
                                                 const float* __restrict__ mask) {
  __shared__ float band[2 * WIN + 1];
  __shared__ float red[256];
  const int gid = blockIdx.x;  // (b*NH+h)*T + t
  const int t = gid % T;
  const int bh = gid / T;
  const int hh = bh % NH, b = bh / NH;
  float* row = scores + (size_t)gid * T;
  const int tid = threadIdx.x;

  if (tid < 2 * WIN + 1) {
    float s = 0.f;
    for (int d = 0; d < HEAD; ++d)
      s += q[((size_t)b * H + hh * HEAD + d) * T + t] * ek[tid * HEAD + d];
    band[tid] = s * 0.125f;
  }
  __syncthreads();

  const float mt = mask[b * T + t];
  float lmax = -1e30f;
  for (int s = tid; s < T; s += 256) {
    float v = row[s];
    int dlt = s - t;
    if (dlt >= -WIN && dlt <= WIN) v += band[dlt + WIN];
    if (mt * mask[b * T + s] == 0.f) v = -10000.f;
    row[s] = v;
    lmax = fmaxf(lmax, v);
  }
  red[tid] = lmax;
  __syncthreads();
  for (int off = 128; off > 0; off >>= 1) {
    if (tid < off) red[tid] = fmaxf(red[tid], red[tid + off]);
    __syncthreads();
  }
  const float mx = red[0];
  __syncthreads();
  float lsum = 0.f;
  for (int s = tid; s < T; s += 256) {
    float e = __expf(row[s] - mx);
    row[s] = e;
    lsum += e;
  }
  red[tid] = lsum;
  __syncthreads();
  for (int off = 128; off > 0; off >>= 1) {
    if (tid < off) red[tid] += red[tid + off];
    __syncthreads();
  }
  const float inv = 1.f / red[0];
  for (int s = tid; s < T; s += 256) row[s] *= inv;
}

// ======================================================================
// ao[b, h*64+d, t] = sum_s v[d,s] * p[t,s]   (one wave per 16x16 tile)
// ======================================================================
__global__ __launch_bounds__(32) void k_attnout(const float* __restrict__ p,
                                                const float* __restrict__ v,
                                                float* __restrict__ ao) {
  alignas(32) __shared__ _Float16 Av[16][48];
  alignas(32) __shared__ _Float16 Bp[16][48];
  const int lane = threadIdx.x;
  const int bh = blockIdx.z;
  const int b = bh / NH, hh = bh % NH;
  const int dBase = blockIdx.x * 16, tBase = blockIdx.y * 16;
  const size_t voff = ((size_t)b * H + hh * HEAD + dBase) * T;

  v8f acc = {};
  for (int s0 = 0; s0 < T; s0 += 32) {
#pragma unroll 4
    for (int j = 0; j < 16; ++j) {
      int idx = j * 32 + lane;  // 16 x 32
      int r = idx >> 5, kk = idx & 31;
      Av[r][kk] = (_Float16)v[voff + (size_t)r * T + s0 + kk];
      Bp[r][kk]       = (_Float16)p[((size_t)bh * T + tBase + r) * T + s0 + kk];
    }
    __syncthreads();
    const int row = lane & 15, koff = (lane >> 4) << 4;
    v16h a  = *(const v16h*)&Av[row][koff];
    v16h bf = *(const v16h*)&Bp[row][koff];
    acc = wmma_f16(a, bf, acc);
    __syncthreads();
  }
  const int hi = lane >> 4, nloc = lane & 15;
#pragma unroll
  for (int r = 0; r < 8; ++r)
    ao[voff + (size_t)(r + hi * 8) * T + tBase + nloc] = acc[r];
}

// ao += band-limited rel-v term: sum_{|d|<=WIN} p[t,t+d] * ev[d+WIN,:]
__global__ void k_relv(const float* __restrict__ p, const float* __restrict__ ev,
                       float* __restrict__ ao) {
  int idx = blockIdx.x * 256 + threadIdx.x;  // over B*H*T
  if (idx >= (int)(BATCH * H * T)) return;
  int t = idx % T, c = (idx / T) % H, b = idx / (H * T);
  int hh = c / HEAD, d = c % HEAD;
  const float* prow = p + ((size_t)(b * NH + hh) * T + t) * T;
  float acc = 0.f;
#pragma unroll
  for (int dl = -WIN; dl <= WIN; ++dl) {
    int s = t + dl;
    if (s >= 0 && s < T) acc += prow[s] * ev[(dl + WIN) * HEAD + d];
  }
  ao[idx] += acc;
}

// h = LayerNorm_channel(h + y*[ymask]) * g + b     (one block per (b,t))
__global__ __launch_bounds__(256) void k_addln(float* __restrict__ h,
                                               const float* __restrict__ y,
                                               const float* __restrict__ g,
                                               const float* __restrict__ bb,
                                               const float* __restrict__ ymask) {
  __shared__ float zbuf[H];
  __shared__ float red[256];
  const int col = blockIdx.x;  // b*T + t
  const int b = col / T, t = col % T;
  const int tid = threadIdx.x;
  const float m = ymask ? ymask[col] : 1.f;

  float lsum = 0.f;
  for (int c = tid; c < H; c += 256) {
    size_t i = ((size_t)b * H + c) * T + t;
    float z = h[i] + y[i] * m;
    zbuf[c] = z;
    lsum += z;
  }
  red[tid] = lsum;
  __syncthreads();
  for (int off = 128; off > 0; off >>= 1) {
    if (tid < off) red[tid] += red[tid + off];
    __syncthreads();
  }
  const float mean = red[0] / (float)H;
  __syncthreads();
  float lv = 0.f;
  for (int c = tid; c < H; c += 256) {
    float d = zbuf[c] - mean;
    lv += d * d;
  }
  red[tid] = lv;
  __syncthreads();
  for (int off = 128; off > 0; off >>= 1) {
    if (tid < off) red[tid] += red[tid + off];
    __syncthreads();
  }
  const float inv = rsqrtf(red[0] / (float)H + 1e-5f);
  for (int c = tid; c < H; c += 256) {
    size_t i = ((size_t)b * H + c) * T + t;
    h[i] = (zbuf[c] - mean) * inv * g[c] + bb[c];
  }
}

__global__ void k_maskmul(float* __restrict__ o, const float* __restrict__ x,
                          const float* __restrict__ mask) {
  int i = blockIdx.x * 256 + threadIdx.x;
  if (i >= (int)(BATCH * H * T)) return;
  int t = i % T, b = i / (H * T);
  o[i] = x[i] * mask[b * T + t];
}

// split prior[B,2H,T] -> out = concat(mu[B,H,T], logs[B,H,T]) with mask
__global__ void k_split(const float* __restrict__ prior,
                        const float* __restrict__ mask, float* __restrict__ out) {
  int i = blockIdx.x * 256 + threadIdx.x;  // over B*H*T
  if (i >= (int)(BATCH * H * T)) return;
  int t = i % T, c = (i / T) % H, b = i / (H * T);
  float mm = mask[b * T + t];
  out[i]                         = prior[((size_t)b * 2 * H + c) * T + t] * mm;
  out[(size_t)BATCH * H * T + i] = prior[((size_t)b * 2 * H + H + c) * T + t] * mm;
}

// ======================================================================
extern "C" void kernel_launch(void* const* d_in, const int* in_sizes, int n_in,
                              void* d_out, int out_size, void* d_ws, size_t ws_size,
                              hipStream_t stream) {
  (void)in_sizes; (void)n_in; (void)out_size; (void)ws_size;
  const float* x   = (const float*)d_in[0];
  const float* xm  = (const float*)d_in[1];
  const float* Wq  = (const float*)d_in[2];
  const float* bq  = (const float*)d_in[3];
  const float* Wk  = (const float*)d_in[4];
  const float* bk  = (const float*)d_in[5];
  const float* Wv  = (const float*)d_in[6];
  const float* bv  = (const float*)d_in[7];
  const float* Wo  = (const float*)d_in[8];
  const float* bo  = (const float*)d_in[9];
  const float* ek  = (const float*)d_in[10];
  const float* ev  = (const float*)d_in[11];
  const float* g1  = (const float*)d_in[12];
  const float* b1  = (const float*)d_in[13];
  const float* g2  = (const float*)d_in[14];
  const float* b2  = (const float*)d_in[15];
  const float* Wf1 = (const float*)d_in[16];
  const float* bf1 = (const float*)d_in[17];
  const float* Wf2 = (const float*)d_in[18];
  const float* bf2 = (const float*)d_in[19];
  const float* Wp  = (const float*)d_in[20];
  const float* bp  = (const float*)d_in[21];

  float* ws = (float*)d_ws;
  size_t off = 0;
  float* h   = ws + off; off += BHT;
  float* qb  = ws + off; off += BHT;
  float* kb  = ws + off; off += BHT;
  float* vb  = ws + off; off += BHT;
  float* ao  = ws + off; off += BHT;
  float* yb  = ws + off; off += BHT;
  float* sc  = ws + off; off += (size_t)BATCH * NH * T * T;
  float* mid = ws + off; off += (size_t)BATCH * FF * T;
  float* pri = ws + off; off += (size_t)BATCH * 2 * H * T;

  // f16 weight arenas (tap-major), carved after the f32 region
  _Float16* hw = (_Float16*)(ws + off);
  size_t ho = 0;
  _Float16* wq16  = hw + ho; ho += (size_t)NL * H * H;
  _Float16* wk16  = hw + ho; ho += (size_t)NL * H * H;
  _Float16* wv16  = hw + ho; ho += (size_t)NL * H * H;
  _Float16* wo16  = hw + ho; ho += (size_t)NL * H * H;
  _Float16* wf116 = hw + ho; ho += (size_t)NL * KS * FF * H;
  _Float16* wf216 = hw + ho; ho += (size_t)NL * KS * H * FF;
  _Float16* wp16  = hw + ho; ho += (size_t)2 * H * H;

  // ---- per-launch weight conversion (deterministic, graph-safe) ----
  auto conv = [&](const float* s, _Float16* d, int M_, int K_, int TP) {
    size_t tot = (size_t)TP * M_ * K_;
    k_w2h<<<(unsigned)((tot + 255) / 256), 256, 0, stream>>>(s, d, M_, K_, TP);
  };
  for (int L = 0; L < NL; ++L) {
    conv(Wq  + (size_t)L * H * H,       wq16  + (size_t)L * H * H,       H,  H,  1);
    conv(Wk  + (size_t)L * H * H,       wk16  + (size_t)L * H * H,       H,  H,  1);
    conv(Wv  + (size_t)L * H * H,       wv16  + (size_t)L * H * H,       H,  H,  1);
    conv(Wo  + (size_t)L * H * H,       wo16  + (size_t)L * H * H,       H,  H,  1);
    conv(Wf1 + (size_t)L * FF * H * KS, wf116 + (size_t)L * KS * FF * H, FF, H,  KS);
    conv(Wf2 + (size_t)L * H * FF * KS, wf216 + (size_t)L * KS * H * FF, H,  FF, KS);
  }
  conv(Wp, wp16, 2 * H, H, 1);

  const int n1 = (BATCH * H * T + 255) / 256;
  k_maskmul<<<n1, 256, 0, stream>>>(h, x, xm);

  for (int L = 0; L < NL; ++L) {
    const float* ekL = ek + (size_t)L * (2 * WIN + 1) * HEAD;
    const float* evL = ev + (size_t)L * (2 * WIN + 1) * HEAD;

    // ---- attention ----
    k_gemm<1, false><<<dim3(H / 64, NTOT / 128), 256, 0, stream>>>(
        wq16 + (size_t)L * H * H, bq + L * H, h, nullptr, qb, H, H);
    k_gemm<1, false><<<dim3(H / 64, NTOT / 128), 256, 0, stream>>>(
        wk16 + (size_t)L * H * H, bk + L * H, h, nullptr, kb, H, H);
    k_gemm<1, false><<<dim3(H / 64, NTOT / 128), 256, 0, stream>>>(
        wv16 + (size_t)L * H * H, bv + L * H, h, nullptr, vb, H, H);
    k_scores<<<dim3(T / 16, T / 16, BATCH * NH), 32, 0, stream>>>(qb, kb, sc);
    k_softmax<<<dim3(BATCH * NH * T), 256, 0, stream>>>(sc, qb, ekL, xm);
    k_attnout<<<dim3(HEAD / 16, T / 16, BATCH * NH), 32, 0, stream>>>(sc, vb, ao);
    k_relv<<<n1, 256, 0, stream>>>(sc, evL, ao);
    k_gemm<1, false><<<dim3(H / 64, NTOT / 128), 256, 0, stream>>>(
        wo16 + (size_t)L * H * H, bo + L * H, ao, nullptr, yb, H, H);
    k_addln<<<dim3(NTOT), 256, 0, stream>>>(h, yb, g1 + L * H, b1 + L * H, nullptr);

    // ---- conv FFN ----
    k_gemm<3, true><<<dim3(FF / 64, NTOT / 128), 256, 0, stream>>>(
        wf116 + (size_t)L * KS * FF * H, bf1 + L * FF, h, xm, mid, FF, H);
    k_gemm<3, false><<<dim3(H / 64, NTOT / 128), 256, 0, stream>>>(
        wf216 + (size_t)L * KS * H * FF, bf2 + L * H, mid, xm, yb, H, FF);
    k_addln<<<dim3(NTOT), 256, 0, stream>>>(h, yb, g2 + L * H, b2 + L * H, xm);
  }

  // ---- prior projection + split ----
  k_gemm<1, false><<<dim3(2 * H / 64, NTOT / 128), 256, 0, stream>>>(
      wp16, bp, h, xm, pri, 2 * H, H);
  k_split<<<n1, 256, 0, stream>>>(pri, xm, (float*)d_out);
}
